// QuantLinear_22600117911583
// MI455X (gfx1250) — matrix operand order repaired and checked by reading, not compiled, not run on hardware
//
#include <hip/hip_runtime.h>

// ---------------------------------------------------------------------------
// NF4 blockwise-dequant fused GEMM for MI455X (gfx1250, wave32, WMMA)
//   out[n, o] = sum_k x[n,k] * (codebook[codes[o,k]] * absmax[(o*I+k)/64])
// Dequant -> f16 tiles in LDS -> v_wmma_f32_16x16x32_f16, f32 accumulate.
// ---------------------------------------------------------------------------

typedef __attribute__((ext_vector_type(16))) _Float16 v16h;
typedef __attribute__((ext_vector_type(8)))  _Float16 v8h;
typedef __attribute__((ext_vector_type(8)))  float    v8f;

#define IN_F   4096
#define OUT_F  16384
#define N_TOK  4096

constexpr int BM  = 128;          // M tile (tokens)
constexpr int BN  = 128;          // N tile (out features)
constexpr int BK  = 64;           // K stage (== bnb block size -> 1 absmax/stage)
constexpr int PAD = 8;            // halves; keeps rows 16B-aligned, cuts bank conflicts
constexpr int LDK = BK + PAD;     // LDS row stride in halves

__device__ __forceinline__ float nf4_lut(int code, int cb_bits) {
    // codebook value lives in lane `code` (0..15); wave32 bpermute, index in bytes
    return __builtin_bit_cast(float,
        __builtin_amdgcn_ds_bpermute(code << 2, cb_bits));
}

__device__ __forceinline__ v16h frag_cat(v8h lo, v8h hi) {
    return __builtin_shufflevector(lo, hi, 0,1,2,3,4,5,6,7,8,9,10,11,12,13,14,15);
}

__global__ __launch_bounds__(256)
void nf4_wmma_gemm(const float* __restrict__ x,
                   const int*   __restrict__ codes,
                   const float* __restrict__ absmax,
                   const float* __restrict__ codebook,
                   float*       __restrict__ out) {
    __shared__ _Float16 sA[BM * LDK];   // x tile,   [m][k] f16
    __shared__ _Float16 sB[BN * LDK];   // W tile,   [o][k] f16 (dequantized)

    const int t    = threadIdx.x;
    const int lane = t & 31;
    const int wave = t >> 5;
    const int wm   = wave >> 2;        // 0..1  -> 64-row M slab
    const int wn   = wave & 3;         // 0..3  -> 32-col N slab
    const int half = lane >> 4;        // hi/lo half-wave
    const int lr   = lane & 15;

    const int m0 = blockIdx.y * BM;
    const int n0 = blockIdx.x * BN;

    // Broadcast the 16-entry NF4 codebook into lanes 0..15 (dup in 16..31).
    const int cb_bits = __builtin_bit_cast(int, codebook[lane & 15]);

    // Stage-load mapping: 2 threads per row, each owning 32 consecutive K.
    const int row = t >> 1;            // 0..127
    const int kh  = (t & 1) * 32;      // 0 or 32

    const float* aRow = x     + (size_t)(m0 + row) * IN_F + kh;
    const int*   cRow = codes + (size_t)(n0 + row) * IN_F + kh;
    const size_t amaxRowBase = ((size_t)(n0 + row) * IN_F) >> 6;

    v8f acc[4][2] = {};   // 4 M-frags x 2 N-frags of 16x16 f32

    for (int ks = 0; ks < IN_F; ks += BK) {
        // ---- stage: global -> (convert/dequant) -> LDS -------------------
        const float4* ap = (const float4*)(aRow + ks);
        const int4*   cp = (const int4*)(cRow + ks);
        // one absmax scalar covers this thread's whole 32-K chunk
        const float scale = absmax[amaxRowBase + (ks >> 6)];

        #pragma unroll
        for (int i = 0; i < 4; ++i) {
            float4 f0 = ap[2 * i], f1 = ap[2 * i + 1];
            v8h ha = {(_Float16)f0.x, (_Float16)f0.y, (_Float16)f0.z, (_Float16)f0.w,
                      (_Float16)f1.x, (_Float16)f1.y, (_Float16)f1.z, (_Float16)f1.w};
            *(v8h*)&sA[row * LDK + kh + i * 8] = ha;

            int4 c0 = cp[2 * i], c1 = cp[2 * i + 1];
            v8h hb;
            hb[0] = (_Float16)(nf4_lut(c0.x, cb_bits) * scale);
            hb[1] = (_Float16)(nf4_lut(c0.y, cb_bits) * scale);
            hb[2] = (_Float16)(nf4_lut(c0.z, cb_bits) * scale);
            hb[3] = (_Float16)(nf4_lut(c0.w, cb_bits) * scale);
            hb[4] = (_Float16)(nf4_lut(c1.x, cb_bits) * scale);
            hb[5] = (_Float16)(nf4_lut(c1.y, cb_bits) * scale);
            hb[6] = (_Float16)(nf4_lut(c1.z, cb_bits) * scale);
            hb[7] = (_Float16)(nf4_lut(c1.w, cb_bits) * scale);
            *(v8h*)&sB[row * LDK + kh + i * 8] = hb;
        }

        // prefetch next stage while this one computes (global_prefetch_b8)
        if (ks + BK < IN_F) {
            __builtin_prefetch(aRow + ks + BK, 0, 1);
            __builtin_prefetch((const char*)(cRow + ks + BK), 0, 1);
            __builtin_prefetch((const char*)(cRow + ks + BK) + 64, 0, 1);
        }

        __syncthreads();

        // ---- compute: 2 k-steps of 32, 8 WMMAs each ----------------------
        #pragma unroll
        for (int k0 = 0; k0 < BK; k0 += 32) {
            v16h afr[4], bfr[2];
            #pragma unroll
            for (int mt = 0; mt < 4; ++mt) {
                // 16-bit A 16x32 layout: lanes 0-15 hold K 0..7 & 16..23,
                // lanes 16-31 hold K 8..15 & 24..31 (ISA 7.12.2)
                const _Float16* p =
                    &sA[(wm * 64 + mt * 16 + lr) * LDK + k0 + half * 8];
                afr[mt] = frag_cat(*(const v8h*)p, *(const v8h*)(p + 16));
            }
            #pragma unroll
            for (int nt = 0; nt < 2; ++nt) {
                // B 32x16: lane = N; lanes 0-15 K 0..7 & 16..23, lanes 16-31 K 8..15 & 24..31
                const _Float16* p =
                    &sB[(wn * 32 + nt * 16 + lr) * LDK + k0 + half * 8];
                bfr[nt] = frag_cat(*(const v8h*)p, *(const v8h*)(p + 16));
            }
            #pragma unroll
            for (int mt = 0; mt < 4; ++mt)
                #pragma unroll
                for (int nt = 0; nt < 2; ++nt)
                    acc[mt][nt] = __builtin_amdgcn_wmma_f32_16x16x32_f16(
                        false, afr[mt], false, bfr[nt],
                        (short)0, acc[mt][nt], false, false);
        }

        __syncthreads();
    }

    // ---- epilogue: C/D layout -> out[n_tok][out_f] f32 -------------------
    #pragma unroll
    for (int mt = 0; mt < 4; ++mt) {
        #pragma unroll
        for (int nt = 0; nt < 2; ++nt) {
            const int gm = m0 + wm * 64 + mt * 16 + half * 8; // + r
            const int gn = n0 + wn * 32 + nt * 16 + lr;
            #pragma unroll
            for (int r = 0; r < 8; ++r)
                out[(size_t)(gm + r) * OUT_F + gn] = acc[mt][nt][r];
        }
    }
}

extern "C" void kernel_launch(void* const* d_in, const int* in_sizes, int n_in,
                              void* d_out, int out_size, void* d_ws, size_t ws_size,
                              hipStream_t stream) {
    const float* x        = (const float*)d_in[0];
    const int*   codes    = (const int*)d_in[1];
    const float* absmax   = (const float*)d_in[2];
    const float* codebook = (const float*)d_in[3];
    float*       out      = (float*)d_out;

    dim3 grid(OUT_F / BN, N_TOK / BM);   // (128, 32)
    dim3 block(256);                     // 8 wave32 waves
    nf4_wmma_gemm<<<grid, block, 0, stream>>>(x, codes, absmax, codebook, out);
}
